// SpatialAttentionModuleLite_1047972020693
// MI455X (gfx1250) — matrix operand order repaired and checked by reading, not compiled, not run on hardware
//
#include <hip/hip_runtime.h>
#include <hip/hip_bf16.h>
#include <math.h>

typedef __attribute__((ext_vector_type(16))) __bf16 v16bf;
typedef __attribute__((ext_vector_type(4)))  __bf16 v4bf;
typedef __attribute__((ext_vector_type(8)))  float  v8f;

#define B_SZ    16
#define N_SEQ   1024
#define H_DIM   512
#define N_HEADS 8
#define HD      64

static __device__ __forceinline__ int lane_id() { return threadIdx.x & 31; }

// ---- WMMA fragment loaders (wave32 layouts per CDNA5 ISA) ----

// A fragment (16x32 bf16): lane half h holds row=lane&15, k = h*16+e
static __device__ __forceinline__ v16bf a_frag_bf16(const __bf16* __restrict__ A, int ld,
                                                    int row0, int k0) {
  const int lane = lane_id();
  const int half = lane >> 4;
  const __bf16* p = A + (size_t)(row0 + (lane & 15)) * ld + k0 + half * 16;
  v16bf a;
#pragma unroll
  for (int e = 0; e < 8; ++e) a[e] = p[e];
#pragma unroll
  for (int e = 0; e < 8; ++e) a[8 + e] = p[8 + e];
  return a;
}

static __device__ __forceinline__ v16bf a_frag_f32(const float* __restrict__ A, int ld,
                                                   int row0, int k0) {
  const int lane = lane_id();
  const int half = lane >> 4;
  const float* p = A + (size_t)(row0 + (lane & 15)) * ld + k0 + half * 16;
  v16bf a;
#pragma unroll
  for (int e = 0; e < 8; ++e) a[e] = (__bf16)p[e];
#pragma unroll
  for (int e = 0; e < 8; ++e) a[8 + e] = (__bf16)p[8 + e];
  return a;
}

// B fragment (32x16 bf16) from row-major f32 (only used in tiny bias kernel)
static __device__ __forceinline__ v16bf b_frag_f32(const float* __restrict__ Bm, int ld,
                                                   int k0, int col0) {
  const int lane = lane_id();
  const int col = col0 + (lane & 15);
  const int kb = k0 + ((lane >> 4) << 4);
  v16bf b;
#pragma unroll
  for (int e = 0; e < 16; ++e) b[e] = (__bf16)Bm[(size_t)(kb + e) * ld + col];
  return b;
}

// B fragment from pre-packed fragment-major weights: one contiguous 32B/lane load.
// Fragment f = ct*numKC + kc holds 512 bf16: element lane*16+e.
static __device__ __forceinline__ v16bf b_frag_packed(const __bf16* __restrict__ P,
                                                      int numKC, int kc, int ct) {
  const __bf16* p = P + ((size_t)(ct * numKC + kc) << 9) + (lane_id() << 4);
  return *(const v16bf*)p;
}

static __device__ __forceinline__ v8f wmma_bf16(v16bf a, v16bf b, v8f c) {
  return __builtin_amdgcn_wmma_f32_16x16x32_bf16(false, a, false, b, (short)0, c,
                                                 false, false);
}

// ---------------- Kernel 0a: f32 -> bf16 stream conversion ----------------
__global__ __launch_bounds__(256) void cvt_bf16_kernel(const float* __restrict__ in,
                                                       __bf16* __restrict__ out, int n4) {
  const int i = blockIdx.x * 256 + threadIdx.x;
  if (i < n4) {
    const float4 f = ((const float4*)in)[i];
    v4bf o;
    o[0] = (__bf16)f.x; o[1] = (__bf16)f.y; o[2] = (__bf16)f.z; o[3] = (__bf16)f.w;
    ((v4bf*)out)[i] = o;
  }
}

// ---------------- Kernel 0b: pack f32 weights into B-fragment-major bf16 --------
__global__ __launch_bounds__(256) void pack_b_kernel(const float* __restrict__ W,
                                                     __bf16* __restrict__ P,
                                                     int K, int N) {
  const int i = blockIdx.x * 256 + threadIdx.x;
  if (i >= K * N) return;
  const int f = i >> 9;
  const int w = i & 511;
  const int lane = w >> 4, e = w & 15;
  const int numKC = K >> 5;
  const int kc = f % numKC, ct = f / numKC;
  const int k = kc * 32 + ((lane >> 4) << 4) + e;
  const int col = ct * 16 + (lane & 15);
  P[i] = (__bf16)W[(size_t)k * N + col];
}

// ---------------- Kernel 1: QKV projection + elu+1 + scale fold ----------------
// Each wave: 16 rows x 64 cols; register double-buffered A/B fragments so each
// WMMA group runs with the next iteration's loads already in flight.
__global__ __launch_bounds__(256) void qkv_kernel(
    const __bf16* __restrict__ xb, const __bf16* __restrict__ Wp,
    const float* __restrict__ bqkv,
    __bf16* __restrict__ qb, __bf16* __restrict__ kbuf, __bf16* __restrict__ vbuf) {
  const int wave = threadIdx.x >> 5;
  const int row0 = blockIdx.x * 16;                 // token tile
  const int ct0 = (blockIdx.y * 8 + wave) * 4;      // first of 4 col tiles
  const int col0 = ct0 * 16;                        // 64-col strip in [0,1536)
  const int lane = lane_id();
  const int half = lane >> 4;
  const int numKC = H_DIM / 32;                     // 16

  v8f acc0 = {}, acc1 = {}, acc2 = {}, acc3 = {};

  v16bf a_cur = a_frag_bf16(xb, H_DIM, row0, 0);
  v16bf b_cur0 = b_frag_packed(Wp, numKC, 0, ct0 + 0);
  v16bf b_cur1 = b_frag_packed(Wp, numKC, 0, ct0 + 1);
  v16bf b_cur2 = b_frag_packed(Wp, numKC, 0, ct0 + 2);
  v16bf b_cur3 = b_frag_packed(Wp, numKC, 0, ct0 + 3);

#pragma unroll 2
  for (int kc = 0; kc < numKC; ++kc) {
    v16bf a_nxt, b_nxt0, b_nxt1, b_nxt2, b_nxt3;
    if (kc + 1 < numKC) {
      a_nxt  = a_frag_bf16(xb, H_DIM, row0, (kc + 1) * 32);
      b_nxt0 = b_frag_packed(Wp, numKC, kc + 1, ct0 + 0);
      b_nxt1 = b_frag_packed(Wp, numKC, kc + 1, ct0 + 1);
      b_nxt2 = b_frag_packed(Wp, numKC, kc + 1, ct0 + 2);
      b_nxt3 = b_frag_packed(Wp, numKC, kc + 1, ct0 + 3);
    }
    acc0 = wmma_bf16(a_cur, b_cur0, acc0);
    acc1 = wmma_bf16(a_cur, b_cur1, acc1);
    acc2 = wmma_bf16(a_cur, b_cur2, acc2);
    acc3 = wmma_bf16(a_cur, b_cur3, acc3);
    a_cur = a_nxt;
    b_cur0 = b_nxt0; b_cur1 = b_nxt1; b_cur2 = b_nxt2; b_cur3 = b_nxt3;
  }

  const int third = col0 >> 9;              // 0=q 1=k 2=v (col0 is 64-aligned)
  const int h = (col0 & 511) >> 6;
  __bf16* dst = (third == 0) ? qb : (third == 1) ? kbuf : vbuf;

#pragma unroll
  for (int t = 0; t < 4; ++t) {
    const v8f acc = (t == 0) ? acc0 : (t == 1) ? acc1 : (t == 2) ? acc2 : acc3;
    const int d = t * 16 + (lane & 15);
    const float bias = bqkv[col0 + t * 16 + (lane & 15)];
#pragma unroll
    for (int r = 0; r < 8; ++r) {
      const int tok = row0 + half * 8 + r;
      const int b_ = tok >> 10;
      const int n = tok & (N_SEQ - 1);
      float v = acc[r] + bias;
      if (third < 2) v = (v > 0.0f) ? (v + 1.0f) : __expf(v);  // elu(x)+1
      if (third == 0) v *= 0.125f;                             // fold 1/sqrt(hd)
      dst[((size_t)(b_ * N_HEADS + h) * N_SEQ + n) * HD + d] = (__bf16)v;
    }
  }
}

// ---------------- Kernel 2: low-rank bias = u @ v  (1024x1024 f32) ----------------
__global__ __launch_bounds__(256) void lowrank_bias_kernel(
    const float* __restrict__ u, const float* __restrict__ v,
    float* __restrict__ biasws) {
  const int wave = threadIdx.x >> 5;
  const int row0 = blockIdx.x * 16;
  const int col0 = (blockIdx.y * 8 + wave) * 16;
  const int lane = lane_id();
  const int half = lane >> 4;
  v8f acc = {};
  acc = wmma_bf16(a_frag_f32(u, 32, row0, 0), b_frag_f32(v, N_SEQ, 0, col0), acc);
  const int col = col0 + (lane & 15);
#pragma unroll
  for (int r = 0; r < 8; ++r)
    biasws[(size_t)(row0 + half * 8 + r) * N_SEQ + col] = acc[r];
}

// ---------------- Kernel 3: flash attention with low-rank bias ----------------
__global__ __launch_bounds__(256) void attn_kernel(
    const __bf16* __restrict__ qb, const __bf16* __restrict__ kb,
    const __bf16* __restrict__ vb, const float* __restrict__ biasws,
    __bf16* __restrict__ ob) {
  __shared__ __bf16 Ks[32 * HD];
  __shared__ __bf16 Vs[32 * HD];
  __shared__ __bf16 Ps[8][16 * 32];

  const int wave = threadIdx.x >> 5;
  const int lane = lane_id();
  const int half = lane >> 4;
  const int bh = blockIdx.x;
  const int n0 = blockIdx.y * 128 + wave * 16;
  const size_t base = (size_t)bh * N_SEQ * HD;

  // Q fragments for the two 32-wide d chunks (1/sqrt(hd) already folded in)
  v16bf qa0 = a_frag_bf16(qb + base, HD, n0, 0);
  v16bf qa1 = a_frag_bf16(qb + base, HD, n0, 32);

  v8f O0 = {}, O1 = {}, O2 = {}, O3 = {};
  float m[8], l[8];
#pragma unroll
  for (int r = 0; r < 8; ++r) { m[r] = -3.0e38f; l[r] = 0.0f; }

  for (int kc = 0; kc < N_SEQ; kc += 32) {
    // cooperative stage of K/V chunk (32x64 bf16 each = 4KB each), b128 copies
    ((uint4*)Ks)[threadIdx.x] = ((const uint4*)(kb + base + (size_t)kc * HD))[threadIdx.x];
    ((uint4*)Vs)[threadIdx.x] = ((const uint4*)(vb + base + (size_t)kc * HD))[threadIdx.x];
    __syncthreads();

    // prefetch next chunk while this one is consumed (global_prefetch_b8)
    if (kc + 32 < N_SEQ) {
      __builtin_prefetch(kb + base + (size_t)(kc + 32) * HD + threadIdx.x * 8, 0, 0);
      __builtin_prefetch(vb + base + (size_t)(kc + 32) * HD + threadIdx.x * 8, 0, 0);
    }

    // S = Q K^T for 32 keys (two 16-col output tiles, two 32-deep d chunks)
    v8f S0 = {}, S1 = {};
#pragma unroll
    for (int dc = 0; dc < 2; ++dc) {
      const int kbase = half << 4;
      v16bf bk0, bk1;
#pragma unroll
      for (int e = 0; e < 16; ++e) {
        bk0[e] = Ks[(lane & 15) * HD + dc * 32 + kbase + e];
        bk1[e] = Ks[((lane & 15) + 16) * HD + dc * 32 + kbase + e];
      }
      v16bf qa = dc ? qa1 : qa0;
      S0 = wmma_bf16(qa, bk0, S0);
      S1 = wmma_bf16(qa, bk1, S1);
    }

    // bias add + online softmax (rows live in one VGPR across a 16-lane half)
#pragma unroll
    for (int r = 0; r < 8; ++r) {
      const int qrow = n0 + half * 8 + r;
      float s0 = S0[r] + biasws[(size_t)qrow * N_SEQ + kc + (lane & 15)];
      float s1 = S1[r] + biasws[(size_t)qrow * N_SEQ + kc + 16 + (lane & 15)];
      float rmax = fmaxf(s0, s1);
#pragma unroll
      for (int j = 8; j >= 1; j >>= 1) rmax = fmaxf(rmax, __shfl_xor(rmax, j, 16));
      const float mnew = fmaxf(m[r], rmax);
      const float scale = __expf(m[r] - mnew);
      const float p0 = __expf(s0 - mnew);
      const float p1 = __expf(s1 - mnew);
      float rsum = p0 + p1;
#pragma unroll
      for (int j = 8; j >= 1; j >>= 1) rsum += __shfl_xor(rsum, j, 16);
      l[r] = l[r] * scale + rsum;
      m[r] = mnew;
      O0[r] *= scale; O1[r] *= scale; O2[r] *= scale; O3[r] *= scale;
      Ps[wave][(half * 8 + r) * 32 + (lane & 15)] = (__bf16)p0;
      Ps[wave][(half * 8 + r) * 32 + 16 + (lane & 15)] = (__bf16)p1;
    }

    // O += P V : A = P (16x32) via per-wave LDS transpose, B = V chunk (32x64)
    v16bf pa = a_frag_bf16(&Ps[wave][0], 32, 0, 0);
#pragma unroll
    for (int t = 0; t < 4; ++t) {
      const int kbase = half << 4;
      v16bf bv;
#pragma unroll
      for (int e = 0; e < 16; ++e)
        bv[e] = Vs[(kbase + e) * HD + t * 16 + (lane & 15)];
      v8f& O = (t == 0) ? O0 : (t == 1) ? O1 : (t == 2) ? O2 : O3;
      O = wmma_bf16(pa, bv, O);
    }
    __syncthreads();
  }

  // normalize and write back to token-major [B*N, H] bf16
  const int b_ = bh >> 3, h = bh & 7;
#pragma unroll
  for (int r = 0; r < 8; ++r) {
    const float inv = 1.0f / l[r];
    const size_t rowoff =
        ((size_t)b_ * N_SEQ + n0 + half * 8 + r) * H_DIM + h * HD;
    ob[rowoff + 0  + (lane & 15)] = (__bf16)(O0[r] * inv);
    ob[rowoff + 16 + (lane & 15)] = (__bf16)(O1[r] * inv);
    ob[rowoff + 32 + (lane & 15)] = (__bf16)(O2[r] * inv);
    ob[rowoff + 48 + (lane & 15)] = (__bf16)(O3[r] * inv);
  }
}

// ---------------- Kernel 4: output projection (16x64 per wave, pipelined) -------
__global__ __launch_bounds__(256) void out_kernel(
    const __bf16* __restrict__ ain, const __bf16* __restrict__ Wp,
    const float* __restrict__ bout, float* __restrict__ out) {
  const int wave = threadIdx.x >> 5;
  const int row0 = blockIdx.x * 16;
  const int ct0 = wave * 4;                         // 8 waves cover all 32 col tiles
  const int col0 = ct0 * 16;
  const int lane = lane_id();
  const int half = lane >> 4;
  const int numKC = H_DIM / 32;                     // 16

  v8f acc0 = {}, acc1 = {}, acc2 = {}, acc3 = {};

  v16bf a_cur = a_frag_bf16(ain, H_DIM, row0, 0);
  v16bf b_cur0 = b_frag_packed(Wp, numKC, 0, ct0 + 0);
  v16bf b_cur1 = b_frag_packed(Wp, numKC, 0, ct0 + 1);
  v16bf b_cur2 = b_frag_packed(Wp, numKC, 0, ct0 + 2);
  v16bf b_cur3 = b_frag_packed(Wp, numKC, 0, ct0 + 3);

#pragma unroll 2
  for (int kc = 0; kc < numKC; ++kc) {
    v16bf a_nxt, b_nxt0, b_nxt1, b_nxt2, b_nxt3;
    if (kc + 1 < numKC) {
      a_nxt  = a_frag_bf16(ain, H_DIM, row0, (kc + 1) * 32);
      b_nxt0 = b_frag_packed(Wp, numKC, kc + 1, ct0 + 0);
      b_nxt1 = b_frag_packed(Wp, numKC, kc + 1, ct0 + 1);
      b_nxt2 = b_frag_packed(Wp, numKC, kc + 1, ct0 + 2);
      b_nxt3 = b_frag_packed(Wp, numKC, kc + 1, ct0 + 3);
    }
    acc0 = wmma_bf16(a_cur, b_cur0, acc0);
    acc1 = wmma_bf16(a_cur, b_cur1, acc1);
    acc2 = wmma_bf16(a_cur, b_cur2, acc2);
    acc3 = wmma_bf16(a_cur, b_cur3, acc3);
    a_cur = a_nxt;
    b_cur0 = b_nxt0; b_cur1 = b_nxt1; b_cur2 = b_nxt2; b_cur3 = b_nxt3;
  }

#pragma unroll
  for (int t = 0; t < 4; ++t) {
    const v8f acc = (t == 0) ? acc0 : (t == 1) ? acc1 : (t == 2) ? acc2 : acc3;
    const int col = col0 + t * 16 + (lane & 15);
    const float bb = bout[col];
#pragma unroll
    for (int r = 0; r < 8; ++r)
      out[(size_t)(row0 + half * 8 + r) * H_DIM + col] = acc[r] + bb;
  }
}

extern "C" void kernel_launch(void* const* d_in, const int* in_sizes, int n_in,
                              void* d_out, int out_size, void* d_ws, size_t ws_size,
                              hipStream_t stream) {
  const float* x    = (const float*)d_in[0];
  const float* Wqkv = (const float*)d_in[1];
  const float* bqkv = (const float*)d_in[2];
  const float* Wout = (const float*)d_in[3];
  const float* bout = (const float*)d_in[4];
  const float* u    = (const float*)d_in[5];
  const float* v    = (const float*)d_in[6];

  char* ws = (char*)d_ws;
  __bf16* qb    = (__bf16*)(ws);                    // 16 MB
  __bf16* kb    = (__bf16*)(ws + (16u << 20));      // 16 MB
  __bf16* vb    = (__bf16*)(ws + (32u << 20));      // 16 MB
  float*  bws   = (float*)(ws + (48u << 20));       // 4 MB  (u@v bias)
  __bf16* attn  = (__bf16*)(ws + (52u << 20));      // 16 MB (attention output)
  __bf16* xb    = (__bf16*)(ws + (68u << 20));      // 16 MB (x in bf16)
  __bf16* Wqkvp = (__bf16*)(ws + (84u << 20));      // 1.5 MB (packed B-frags)
  __bf16* Woutp = (__bf16*)(ws + (86u << 20));      // 0.5 MB (packed B-frags)

  // one-time conversions / weight packing
  const int nx4 = (B_SZ * N_SEQ * H_DIM) / 4;       // 2,097,152
  cvt_bf16_kernel<<<(nx4 + 255) / 256, 256, 0, stream>>>(x, xb, nx4);
  const int nwq = H_DIM * 3 * H_DIM;                // 786,432
  const int nwo = H_DIM * H_DIM;                    // 262,144
  pack_b_kernel<<<(nwq + 255) / 256, 256, 0, stream>>>(Wqkv, Wqkvp, H_DIM, 3 * H_DIM);
  pack_b_kernel<<<(nwo + 255) / 256, 256, 0, stream>>>(Wout, Woutp, H_DIM, H_DIM);

  // QKV: 1024 row tiles x 24 col strips (8 strips/block)
  qkv_kernel<<<dim3(1024, 3), 256, 0, stream>>>(xb, Wqkvp, bqkv, qb, kb, vb);
  // bias: 64 x 64 tiles
  lowrank_bias_kernel<<<dim3(64, 8), 256, 0, stream>>>(u, v, bws);
  // attention: 128 (b,h) x 8 row blocks
  attn_kernel<<<dim3(128, 8), 256, 0, stream>>>(qb, kb, vb, bws, attn);
  // output projection: 1024 row tiles, 8 col strips per block
  out_kernel<<<dim3(1024, 1), 256, 0, stream>>>(attn, Woutp, bout, (float*)d_out);
}
